// RelInteraction_27032524161361
// MI455X (gfx1250) — matrix operand order repaired and checked by reading, not compiled, not run on hardware
//
#include <hip/hip_runtime.h>
#include <cstdint>
#include <cstddef>

typedef __bf16 bf16;
typedef __attribute__((ext_vector_type(16))) __bf16 bf16x16;
typedef __attribute__((ext_vector_type(8)))  float  f32x8;
typedef __attribute__((ext_vector_type(4)))  float  f32x4;
typedef __attribute__((ext_vector_type(4))) unsigned int u32x4;
typedef __attribute__((ext_vector_type(8))) int i32x8;
typedef __attribute__((ext_vector_type(4))) int i32x4;

constexpr int NIM   = 2;
constexpr int NOBJ  = 64;
constexpr int NPAIR = 256;
constexpr int C_    = 512;
constexpr int H_    = 28;
constexpr int W_    = 28;
constexpr int HW_   = H_ * W_;     // 784
constexpr int POOL_ = 7;
constexpr int PP_   = 49;
constexpr int PDIM_ = 4096;
constexpr int FLAT_ = C_ * PP_;    // 25088
constexpr int NREL_ = 51;

// ---------------------------------------------------------------------------
__global__ __launch_bounds__(256) void k_cvt_bf16(const float* __restrict__ in,
                                                  bf16* __restrict__ out, int n) {
  int i = blockIdx.x * 256 + threadIdx.x;
  if (i < n) out[i] = (bf16)in[i];
}

// NCHW fp32 -> NHWC bf16 (fmaps: 2 x 512 x 784 -> 2 x 784 x 512)
__global__ __launch_bounds__(256) void k_nchw_to_nhwc(const float* __restrict__ in,
                                                      bf16* __restrict__ out) {
  const int hw = blockIdx.x, b = blockIdx.y;
  const int c = threadIdx.x;
  const float* src = in + (long)b * C_ * HW_ + hw;
  bf16* dst = out + ((long)(b * HW_ + hw)) * C_;
  dst[c]       = (bf16)src[(long)c * HW_];
  dst[c + 256] = (bf16)src[(long)(c + 256) * HW_];
}

// Per-object 3x3 depthwise conv with generated filters, NHWC bf16 in/out.
__global__ __launch_bounds__(256) void k_depthwise(const bf16* __restrict__ red,
                                                   const float* __restrict__ filt,
                                                   const float* __restrict__ rois,
                                                   bf16* __restrict__ fd) {
  const int hw = blockIdx.x, n = blockIdx.y;
  const int h = hw / W_, w = hw % W_;
  const int im = (int)rois[n * 5];
  const int c0 = threadIdx.x, c1 = threadIdx.x + 256;
  const float* fb = filt + (long)n * (C_ * 9);
  float f0[9], f1[9];
#pragma unroll
  for (int t = 0; t < 9; ++t) { f0[t] = fb[c0 * 9 + t]; f1[t] = fb[c1 * 9 + t]; }
  float a0 = 0.f, a1 = 0.f;
#pragma unroll
  for (int i = 0; i < 3; ++i) {
    const int y = h + i - 1;
    if (y < 0 || y >= H_) continue;
#pragma unroll
    for (int j = 0; j < 3; ++j) {
      const int x = w + j - 1;
      if (x < 0 || x >= W_) continue;
      const bf16* r = red + ((long)(im * HW_ + y * W_ + x)) * C_;
      a0 += (float)r[c0] * f0[i * 3 + j];
      a1 += (float)r[c1] * f1[i * 3 + j];
    }
  }
  bf16* d = fd + ((long)(n * HW_ + hw)) * C_;
  d[c0] = (bf16)a0;
  d[c1] = (bf16)a1;
}

// ROI-align (bilinear, 7x7) over NHWC maps -> pooled (256, 49, 1024) bf16
__global__ __launch_bounds__(256) void k_roi_align(const bf16* __restrict__ fp,
                                                   const float* __restrict__ rois,
                                                   const int* __restrict__ rel,
                                                   bf16* __restrict__ pooled) {
  const int hw = blockIdx.x;      // i*7+j (i over y, j over x)
  const int which = blockIdx.y;   // 0 = sub, 1 = obj
  const int p = blockIdx.z;
  const int i = hw / POOL_, j = hw % POOL_;
  const int obj = rel[p * 3 + 1 + which];
  const float s = 1.0f / 16.0f;
  const float bx0 = rois[obj * 5 + 1] * s, by0 = rois[obj * 5 + 2] * s;
  const float bx1 = rois[obj * 5 + 3] * s, by1 = rois[obj * 5 + 4] * s;
  const float gx = (j + 0.5f) / POOL_, gy = (i + 0.5f) / POOL_;
  float xs = bx0 + gx * (bx1 - bx0); xs = fminf(fmaxf(xs, 0.f), (float)(W_ - 1));
  float ys = by0 + gy * (by1 - by0); ys = fminf(fmaxf(ys, 0.f), (float)(H_ - 1));
  const int x0 = (int)floorf(xs), y0 = (int)floorf(ys);
  const int x1 = min(x0 + 1, W_ - 1), y1 = min(y0 + 1, H_ - 1);
  const float wx = xs - (float)x0, wy = ys - (float)y0;
  const bf16* base = fp + (long)obj * HW_ * C_;
  const bf16* p00 = base + (long)(y0 * W_ + x0) * C_;
  const bf16* p01 = base + (long)(y0 * W_ + x1) * C_;
  const bf16* p10 = base + (long)(y1 * W_ + x0) * C_;
  const bf16* p11 = base + (long)(y1 * W_ + x1) * C_;
  bf16* out = pooled + ((long)(p * PP_ + hw)) * (2 * C_) + which * C_;
  for (int c = threadIdx.x; c < C_; c += 256) {
    float v = (float)p00[c] * (1 - wy) * (1 - wx) + (float)p01[c] * (1 - wy) * wx +
              (float)p10[c] * wy * (1 - wx) + (float)p11[c] * wy * wx;
    out[c] = (bf16)v;
  }
}

// ---------------------------------------------------------------------------
// WMMA bf16 GEMM:  D[m][n] = act( sum_k A[m][k] * Wm[n][k] + bias[n] )
// Double-buffered, software-pipelined K loop:
//   A tiles  : global_load_async_to_lds_b128 (ASYNCcnt)
//   W tiles  : fp32 streamed non-temporally (single-use, > L2), converted to
//              bf16 in-register; loads issued before, committed to LDS after,
//              this tile's WMMAs. When W is bf16 (TDMW): one Tensor Data Mover
//              2D descriptor per K-step with pad_enable matching the LDS pitch
//              (TENSORcnt).
template <int BM, int BN, int WM, int WN, typename WT, typename OT,
          bool RELU, bool PERM49, bool NGUARD, bool TDMW>
__global__ __launch_bounds__(256) void k_wmma_gemm(
    const bf16* __restrict__ A, int lda,
    const WT* __restrict__ Wm,
    const float* __restrict__ bias,
    OT* __restrict__ D, int ldd,
    int M, int N, int K,
    long sA, long sW, long sD) {
  constexpr int KT = 32, PAD = 8, LDK = KT + PAD;  // 80B row pitch (16B aligned)
  constexpr int MT = BM / (16 * WM);
  constexpr int NT = BN / (16 * WN);
  constexpr int IT = BN * 8 / 256;  // float4 units per thread for fp32 W staging
  static_assert(WM * WN == 8, "8 waves per block");
  static_assert(!TDMW || sizeof(WT) == 2, "TDM path streams bf16 weights");

  __shared__ __attribute__((aligned(16))) bf16 As[2][BM][LDK];
  __shared__ __attribute__((aligned(16))) bf16 Ws[2][BN][LDK];

  const int tid = threadIdx.x;
  const int wave = tid >> 5, lane = tid & 31;
  const int half = lane >> 4, l16 = lane & 15;
  const int wm = wave % WM, wn = wave / WM;
  const int m0 = blockIdx.y * BM, n0 = blockIdx.x * BN;

  const bf16* Ab = A + (long)blockIdx.z * sA;
  const WT* Wb = Wm + (long)blockIdx.z * sW;
  OT* Db = D + (long)blockIdx.z * sD;

  // ---- stage A tile (BM x 32 bf16): async copies, no VGPR round-trip
  auto stage_a = [&](int k0, int buf) {
#pragma unroll
    for (int t = 0; t < (BM * 4 + 255) / 256; ++t) {
      const int u = tid + t * 256;
      if (u < BM * 4) {
        const int row = u >> 2, col = (u & 3) * 8;
        const uint64_t g = (uint64_t)(Ab + (long)(m0 + row) * lda + k0 + col);
        const uint32_t l = (uint32_t)(uintptr_t)&As[buf][row][col];
        asm volatile("global_load_async_to_lds_b128 %0, %1, off"
                     :: "v"(l), "v"(g) : "memory");
      }
    }
  };

  float4 wreg[IT];  // fp32 W staging registers (live across the WMMA section)

  // ---- W tile phase 1: issue global loads (fp32 path) or TDM descriptor
  auto stage_w_issue = [&](int k0, int buf) {
    if constexpr (!TDMW) {
#pragma unroll
      for (int t = 0; t < IT; ++t) {
        const int u = tid + t * 256;
        const int row = u >> 3;
        if (!NGUARD || n0 + row < N) {
          const f32x4* src =
              (const f32x4*)((const float*)Wb + (long)(n0 + row) * K + k0 + (u & 7) * 4);
          const f32x4 v = __builtin_nontemporal_load(src);  // single-use stream, skip L2 reuse
          wreg[t] = float4{v[0], v[1], v[2], v[3]};
        } else {
          wreg[t] = float4{0.f, 0.f, 0.f, 0.f};
        }
      }
    } else {
      if (wave == 0) {  // one TDM descriptor per block per K-step
        const uint64_t g = (uint64_t)((const bf16*)Wb + (long)n0 * K + k0);
        const uint32_t l = (uint32_t)(uintptr_t)&Ws[buf][0][0];
        const u32x4 g0 = {1u,                                   // count=1
                          l,                                    // lds_addr
                          (unsigned)(g & 0xffffffffu),          // global_addr lo
                          (unsigned)((g >> 32) & 0x01ffffffu) | (2u << 30)};  // hi | type=2
        const i32x8 g1 = {
            (1 << 16) | (1 << 20) | (3 << 22) | (3 << 25),  // 2B elems, pad: +4dw every 16dw
            (K & 0xffff) << 16,                             // tensor_dim0[15:0]
            (K >> 16) | ((N & 0xffff) << 16),               // tensor_dim0 hi | tensor_dim1 lo
            (N >> 16) | (KT << 16),                         // tensor_dim1 hi | tile_dim0=32
            BN,                                             // tile_dim1=128 (tile_dim2=0)
            K,                                              // tensor_dim0_stride
            0, 0};
        const i32x4 g2 = {0, 0, 0, 0};
        const i32x4 g3 = {0, 0, 0, 0};
#if __clang_major__ >= 23
        const i32x8 z8 = {0, 0, 0, 0, 0, 0, 0, 0};
        __builtin_amdgcn_tensor_load_to_lds(g0, g1, g2, g3, z8, 0);
#else
        __builtin_amdgcn_tensor_load_to_lds(g0, g1, g2, g3, 0);
#endif
      }
    }
  };

  // ---- W tile phase 2: convert + commit to LDS (fp32 path only)
  auto stage_w_commit = [&](int buf) {
    if constexpr (!TDMW) {
#pragma unroll
      for (int t = 0; t < IT; ++t) {
        const int u = tid + t * 256;
        const int row = u >> 3, col = (u & 7) * 4;
        bf16* d = &Ws[buf][row][col];
        d[0] = (bf16)wreg[t].x; d[1] = (bf16)wreg[t].y;
        d[2] = (bf16)wreg[t].z; d[3] = (bf16)wreg[t].w;
      }
    }
  };

  f32x8 acc[MT][NT];
#pragma unroll
  for (int i = 0; i < MT; ++i)
#pragma unroll
    for (int j = 0; j < NT; ++j)
#pragma unroll
      for (int e = 0; e < 8; ++e) acc[i][j][e] = 0.0f;

  const int kTiles = K / KT;
  stage_a(0, 0);
  stage_w_issue(0, 0);
  stage_w_commit(0);

  for (int kt = 0; kt < kTiles; ++kt) {
    // publish tile kt (copies were issued one iteration earlier)
    asm volatile("s_wait_asynccnt 0x0" ::: "memory");
    if constexpr (TDMW) {
      if (wave == 0) __builtin_amdgcn_s_wait_tensorcnt(0);
    }
    __syncthreads();
    const int cur = kt & 1;
    const bool pre = (kt + 1 < kTiles);
    if (pre) {  // overlap next tile's copies/loads with this tile's WMMAs
      stage_a((kt + 1) * KT, cur ^ 1);
      stage_w_issue((kt + 1) * KT, cur ^ 1);
    }

    const bf16(*Ac)[LDK] = As[cur];
    const bf16(*Wc)[LDK] = Ws[cur];
    // A frag: lane L -> M=L&15; VGPR 0-3: K=8*half+0..7, VGPR 4-7: K=16+8*half+0..7
    bf16x16 af[MT];
#pragma unroll
    for (int i = 0; i < MT; ++i) {
      const bf16* ap = &Ac[(wm * MT + i) * 16 + l16][0];
      ((uint4*)&af[i])[0] = *(const uint4*)(ap + 8 * half);
      ((uint4*)&af[i])[1] = *(const uint4*)(ap + 16 + 8 * half);
    }
    // B frag: lane L -> N=L&15; VGPR 0-3: K=8*half+0..7, VGPR 4-7: K=16+8*half+0..7
    // MT==1: prefetch all NT fragments so the scheduler can use staggered
    // partial s_wait_dscnt instead of a full wait per WMMA.
    // MT>1 : 2-slot rotation (keeps the 128-VGPR-accumulator config spill-free;
    // each dscnt wait already amortizes over MT WMMAs).
    constexpr int NB = (MT == 1) ? NT : 2;
    bf16x16 bfr[NB];
    auto load_b = [&](int j, int slot) {
      const bf16* bp = &Wc[(wn * NT + j) * 16 + l16][8 * half];
      ((uint4*)&bfr[slot])[0] = *(const uint4*)(bp);
      ((uint4*)&bfr[slot])[1] = *(const uint4*)(bp + 16);
    };
    if constexpr (MT == 1) {
#pragma unroll
      for (int j = 0; j < NT; ++j) load_b(j, j);
#pragma unroll
      for (int j = 0; j < NT; ++j)
        acc[0][j] = __builtin_amdgcn_wmma_f32_16x16x32_bf16(
            false, af[0], false, bfr[j], (short)0, acc[0][j], false, false);
    } else {
      load_b(0, 0);
#pragma unroll
      for (int j = 0; j < NT; ++j) {
        if (j + 1 < NT) load_b(j + 1, (j + 1) & 1);
#pragma unroll
        for (int i = 0; i < MT; ++i)
          acc[i][j] = __builtin_amdgcn_wmma_f32_16x16x32_bf16(
              false, af[i], false, bfr[j & 1], (short)0, acc[i][j], false, false);
      }
    }

    if (pre) stage_w_commit(cur ^ 1);  // loadcnt wait lands after the WMMAs
  }

  // ---- store: C/D layout: lane L -> N=L&15; VGPR r -> M = r + 8*(L>>4)
#pragma unroll
  for (int j = 0; j < NT; ++j) {
    const int n = n0 + (wn * NT + j) * 16 + l16;
    if (n >= N) continue;
    const float bv = bias ? bias[n] : 0.0f;
#pragma unroll
    for (int i = 0; i < MT; ++i) {
#pragma unroll
      for (int r = 0; r < 8; ++r) {
        const int m = m0 + (wm * MT + i) * 16 + r + 8 * half;
        if (m >= M) continue;
        float v = acc[i][j][r] + bv;
        if (RELU) v = fmaxf(v, 0.0f);
        long idx;
        if constexpr (PERM49) {  // m = p*49+hw ; flatten as (p, n*49+hw)
          const int p = m / 49, hw = m - p * 49;
          idx = (long)p * ldd + (long)n * 49 + hw;
        } else {
          idx = (long)m * ldd + n;
        }
        Db[idx] = (OT)v;
      }
    }
  }
}

// ---------------------------------------------------------------------------
extern "C" void kernel_launch(void* const* d_in, const int* in_sizes, int n_in,
                              void* d_out, int out_size, void* d_ws, size_t ws_size,
                              hipStream_t stream) {
  (void)in_sizes; (void)n_in; (void)out_size; (void)ws_size;
  const float* fmaps = (const float*)d_in[0];
  const float* rois  = (const float*)d_in[1];
  const int*   rel   = (const int*)d_in[2];
  const float* feats = (const float*)d_in[3];
  const float* w_dw  = (const float*)d_in[4];
  const float* b_dw  = (const float*)d_in[5];
  const float* w_pw  = (const float*)d_in[6];
  const float* b_pw  = (const float*)d_in[7];
  const float* w_red = (const float*)d_in[8];
  const float* b_red = (const float*)d_in[9];
  const float* w_rec = (const float*)d_in[10];
  const float* b_rec = (const float*)d_in[11];
  const float* w_fc1 = (const float*)d_in[12];
  const float* b_fc1 = (const float*)d_in[13];
  const float* w_fc2 = (const float*)d_in[14];
  const float* b_fc2 = (const float*)d_in[15];
  const float* w_rel = (const float*)d_in[16];
  const float* b_rel = (const float*)d_in[17];
  float* out = (float*)d_out;

  char* ws = (char*)d_ws;
  size_t off = 0;
  auto take = [&](size_t bytes) -> void* {
    off = (off + 255) & ~(size_t)255;
    void* p = ws + off;
    off += bytes;
    return p;
  };
  bf16*  feats_bf = (bf16*)take((size_t)NOBJ * C_ * 2);
  bf16*  fmaps_t  = (bf16*)take((size_t)NIM * HW_ * C_ * 2);
  bf16*  red_t    = (bf16*)take((size_t)NIM * HW_ * C_ * 2);
  float* filt_dw  = (float*)take((size_t)NOBJ * C_ * 9 * 4);
  bf16*  filt_pw  = (bf16*)take((size_t)NOBJ * C_ * C_ * 2);
  bf16*  fd_t     = (bf16*)take((size_t)NOBJ * HW_ * C_ * 2);
  bf16*  fp_t     = (bf16*)take((size_t)NOBJ * HW_ * C_ * 2);
  bf16*  pooled   = (bf16*)take((size_t)NPAIR * PP_ * 2 * C_ * 2);
  bf16*  xflat    = (bf16*)take((size_t)NPAIR * FLAT_ * 2);
  bf16*  x1       = (bf16*)take((size_t)NPAIR * PDIM_ * 2);
  bf16*  x2       = (bf16*)take((size_t)NPAIR * PDIM_ * 2);

  k_cvt_bf16<<<dim3((NOBJ * C_ + 255) / 256), dim3(256), 0, stream>>>(feats, feats_bf, NOBJ * C_);
  k_nchw_to_nhwc<<<dim3(HW_, NIM), dim3(256), 0, stream>>>(fmaps, fmaps_t);

  // 1) filt_dw = feats @ w_dw.T + b_dw            (64 x 4608, K=512)
  k_wmma_gemm<64, 128, 4, 2, float, float, false, false, false, false>
      <<<dim3(36, 1, 1), 256, 0, stream>>>(feats_bf, C_, w_dw, b_dw, filt_dw, C_ * 9,
                                           NOBJ, C_ * 9, C_, 0, 0, 0);
  // 2) filt_pw = feats @ w_pw.T + b_pw            (64 x 262144, K=512) -> bf16
  k_wmma_gemm<64, 128, 4, 2, float, bf16, false, false, false, false>
      <<<dim3(2048, 1, 1), 256, 0, stream>>>(feats_bf, C_, w_pw, b_pw, filt_pw, C_ * C_,
                                             NOBJ, C_ * C_, C_, 0, 0, 0);
  // 3) red = relu(fmaps_nhwc @ w_red.T + b_red)   (1568 x 512, K=512)
  k_wmma_gemm<16, 128, 1, 8, float, bf16, true, false, false, false>
      <<<dim3(4, 98, 1), 256, 0, stream>>>(fmaps_t, C_, w_red, b_red, red_t, C_,
                                           NIM * HW_, C_, C_, 0, 0, 0);
  // 4) per-object depthwise 3x3 (zero pad)
  k_depthwise<<<dim3(HW_, NOBJ), 256, 0, stream>>>(red_t, filt_dw, rois, fd_t);

  // 5) fp[n] = relu(fd[n] @ filt_pw[n].T)         batched (784 x 512, K=512) x 64; TDM weights
  k_wmma_gemm<16, 128, 1, 8, bf16, bf16, true, false, false, true>
      <<<dim3(4, 49, NOBJ), 256, 0, stream>>>(fd_t, C_, filt_pw, nullptr, fp_t, C_,
                                              HW_, C_, C_,
                                              (long)HW_ * C_, (long)C_ * C_, (long)HW_ * C_);
  // 6) ROI-align sub/obj -> pooled (256, 49, 1024)
  k_roi_align<<<dim3(PP_, 2, NPAIR), 256, 0, stream>>>(fp_t, rois, rel, pooled);

  // 7) rec: (12544 x 512, K=1024), scattered store into x[p][d*49+hw]
  k_wmma_gemm<256, 128, 4, 2, float, bf16, false, true, false, false>
      <<<dim3(4, 49, 1), 256, 0, stream>>>(pooled, 2 * C_, w_rec, b_rec, xflat, FLAT_,
                                           NPAIR * PP_, C_, 2 * C_, 0, 0, 0);
  // 8) fc1: relu(x @ w_fc1.T + b)  (256 x 4096, K=25088); grid.y=1 -> w_fc1 streamed once
  k_wmma_gemm<256, 128, 4, 2, float, bf16, true, false, false, false>
      <<<dim3(32, 1, 1), 256, 0, stream>>>(xflat, FLAT_, w_fc1, b_fc1, x1, PDIM_,
                                           NPAIR, PDIM_, FLAT_, 0, 0, 0);
  // 9) fc2: (256 x 4096, K=4096)
  k_wmma_gemm<256, 128, 4, 2, float, bf16, false, false, false, false>
      <<<dim3(32, 1, 1), 256, 0, stream>>>(x1, PDIM_, w_fc2, b_fc2, x2, PDIM_,
                                           NPAIR, PDIM_, PDIM_, 0, 0, 0);
  // 10) head: (256 x 51, K=4096) -> fp32 output (N-guarded)
  k_wmma_gemm<256, 128, 4, 2, float, float, false, false, true, false>
      <<<dim3(1, 1, 1), 256, 0, stream>>>(x2, PDIM_, w_rel, b_rel, out, NREL_,
                                          NPAIR, NREL_, PDIM_, 0, 0, 0);
}
